// KerasALIF_33672543601225
// MI455X (gfx1250) — compile-verified, matrix operand order checked by reading, not run on hardware
//
#include <hip/hip_runtime.h>
#include <hip/hip_bf16.h>

typedef __attribute__((ext_vector_type(16))) _Float16 v16h;
typedef __attribute__((ext_vector_type(8)))  _Float16 v8h;
typedef __attribute__((ext_vector_type(8)))  float    v8f;
typedef __attribute__((ext_vector_type(4)))  float    v4f;

#define T_STEPS 1000
#define NIN     128
#define UNITS   512
#define KTOT    640          // NIN + UNITS
#define KTILES  20           // KTOT / 32
#define NTILES  32           // UNITS / 16

// decay = exp(-1/20), decay_b = exp(-1/200)
#define DECAY    0.95122942450071400f
#define OMDECAY  0.04877057549928600f
#define DECAYB   0.99501247919268232f
#define OMDECAYB 0.00498752080731768f
#define THR0     0.01f
#define BETA     1.6f

// ---- 16-bit WMMA operand layout helpers (ISA 7.12.2) ----
// k_local = (v<4 ? 2v : 16+2(v-4)) + p + 8h  with v=e>>1, p=e&1
__device__ __forceinline__ int kl_from_e(int e, int h) {
    int v = e >> 1, p = e & 1;
    return ((v < 4) ? (2 * v) : (16 + 2 * (v - 4))) + p + 8 * h;
}
// inverse: k_local -> (half, e)
__device__ __forceinline__ void e_from_kl(int kl, int& h, int& e) {
    h = (kl >> 3) & 1;
    int k0 = kl - 8 * h;                       // in {0..7} u {16..23}
    int v = (k0 < 8) ? (k0 >> 1) : (4 + ((k0 - 16) >> 1));
    e = 2 * v + (k0 & 1);
}

// ---- pack [W_in ; W_rec(diag=0)] into f16 WMMA-B tile order ----
// Wp[(kt*NTILES + ntg)*512 + lane*16 + e]
__global__ __launch_bounds__(256) void pack_w_kernel(const float* __restrict__ W_in,
                                                     const float* __restrict__ W_rec,
                                                     _Float16* __restrict__ Wp) {
    int idx = blockIdx.x * 256 + threadIdx.x;
    if (idx >= KTILES * NTILES * 512) return;
    int e    = idx & 15;
    int lane = (idx >> 4) & 31;
    int tile = idx >> 9;            // kt*NTILES + ntg
    int ntg  = tile & (NTILES - 1);
    int kt   = tile >> 5;
    int n    = ntg * 16 + (lane & 15);
    int h    = lane >> 4;
    int k    = kt * 32 + kl_from_e(e, h);
    float val;
    if (k < NIN) {
        val = W_in[(size_t)k * UNITS + n];
    } else {
        int j = k - NIN;
        val = (j == n) ? 0.0f : W_rec[(size_t)j * UNITS + n];
    }
    Wp[idx] = (_Float16)val;
}

// ---- fused ALIF scan: one WG = 16 batch rows, all T steps ----
__global__ __launch_bounds__(256) void alif_scan_kernel(const float* __restrict__ x,
                                                        const _Float16* __restrict__ Wp,
                                                        float* __restrict__ out) {
    // A operand staging: [kt][lane][e] halves; kt 0..3 = x_t, kt 4..19 = z
    __shared__ __align__(32) _Float16 Abuf[KTILES * 32 * 16];   // 20 KB

    const int tid  = threadIdx.x;
    const int wave = tid >> 5;        // 0..7 -> units [wave*64, wave*64+64)
    const int lane = tid & 31;
    const int lm   = lane & 15;
    const int lh   = lane >> 4;
    const int bt   = blockIdx.x;      // batch tile (16 rows)

    // zero the z-region (kt 4..19) before first step
    for (int i = tid; i < (KTILES - 4) * 512; i += 256)
        Abuf[4 * 512 + i] = (_Float16)0.0f;

    // ---------- loop-invariant addressing ----------
    // x staging: thread handles row m, features [i0, i0+8) -> one 16B LDS store
    const int m_st = tid >> 4;                 // 16 threads per row
    const int i0   = (tid & 15) * 8;           // 8-aligned feature base
    const int kt_s = i0 >> 5;
    const int blk  = (i0 & 31) >> 3;           // 0..3
    const int h_s  = blk & 1;                  // blocks 1,3 -> upper half-lanes
    const int e0_s = (blk >> 1) * 8;           // blocks 2,3 -> e base 8
    _Float16* const stg = &Abuf[kt_s * 512 + (m_st + 16 * h_s) * 16 + e0_s];
    const float* xp = x + (size_t)(bt * 16 + m_st) * T_STEPS * NIN + i0;

    // weight base for this lane (per-tile stride handled by immediate offsets)
    const _Float16* const wpl = Wp + (size_t)wave * 4 * 512 + lane * 16;

    // z scatter: per nt fixed (ktz, h2, e2); per g offset = g*16 halves
    _Float16* zdst[4];
#pragma unroll
    for (int nt = 0; nt < 4; ++nt) {
        int n = wave * 64 + nt * 16 + lm;
        int ktz = 4 + (n >> 5);
        int h2, e2; e_from_kl(n & 31, h2, e2);
        zdst[nt] = &Abuf[ktz * 512 + (8 * lh + 16 * h2) * 16 + e2];
    }
    // output base: row (bt*16 + 8*lh + g), col (wave*64 + nt*16 + lm)
    float* const outp = out + (size_t)(bt * 16 + 8 * lh) * T_STEPS * UNITS
                            + wave * 64 + lm;

    // per-lane state in WMMA C-layout
    float vS[4][8], bS[4][8], zS[4][8];
#pragma unroll
    for (int nt = 0; nt < 4; ++nt)
#pragma unroll
        for (int g = 0; g < 8; ++g) { vS[nt][g] = 0.f; bS[nt][g] = 0.f; zS[nt][g] = 0.f; }

    for (int t = 0; t < T_STEPS; ++t) {
        // ---- stage x_t: 2 x v4f NT loads -> 8 halves -> one ds_store_b128
        // x is read exactly once across the whole kernel: non-temporal keeps
        // L2 free for the 1000x-reused weights.
        v4f a0 = __builtin_nontemporal_load((const v4f*)(xp));
        v4f a1 = __builtin_nontemporal_load((const v4f*)(xp + 4));
        xp += NIN;
        __builtin_prefetch(xp, 0, 1);          // next step's row segment
        v8h hv = { (_Float16)a0.x, (_Float16)a0.y, (_Float16)a0.z, (_Float16)a0.w,
                   (_Float16)a1.x, (_Float16)a1.y, (_Float16)a1.z, (_Float16)a1.w };
        *(v8h*)stg = hv;
        __syncthreads();   // x_t staged + z(t-1) visible

        // ---- GEMM: i_t tile [16 x 64] per wave, K = 640
        // Launder a 32-bit OFFSET (not the pointer!) so (a) the compiler cannot
        // hoist the t-invariant weight loads out of the T loop and spill, and
        // (b) pointer provenance to the kernel arg survives -> addrspace(1) is
        // inferred -> global_load_b128 (not flat_load, which also ticks DScnt).
        unsigned toff = 0;
        asm volatile("" : "+v"(toff));
        const _Float16* wl = wpl + toff;

        v8f acc[4] = {};
        for (int kt = 0; kt < KTILES; ++kt) {
            v16h a = *(const v16h*)&Abuf[kt * 512 + lane * 16];
#pragma unroll
            for (int nt = 0; nt < 4; ++nt) {
                v16h bm = *(const v16h*)&wl[(size_t)kt * (NTILES * 512) + nt * 512];
                acc[nt] = __builtin_amdgcn_wmma_f32_16x16x32_f16(
                    false, a, false, bm, (short)0, acc[nt], false, false);
            }
        }
        __syncthreads();   // all A reads done; safe to overwrite z-region

        // ---- elementwise ALIF update, spike, state + output + z scatter
#pragma unroll
        for (int nt = 0; nt < 4; ++nt) {
#pragma unroll
            for (int g = 0; g < 8; ++g) {
                float it = acc[nt][g];                       // x@W_in + z@W_rec
                float zp = zS[nt][g];
                float nb = DECAYB * bS[nt][g] + OMDECAYB * zp;
                float th = THR0 + BETA * nb;
                float nv = DECAY * vS[nt][g] + OMDECAY * it - zp * th;
                float zn = (nv > th) ? 1.0f : 0.0f;          // thr > 0 always
                vS[nt][g] = nv; bS[nt][g] = nb; zS[nt][g] = zn;
                // 524 MB write-once stream: non-temporal, don't pollute L2
                __builtin_nontemporal_store(
                    zn, &outp[(size_t)g * T_STEPS * UNITS + (size_t)t * UNITS + nt * 16]);
                zdst[nt][g * 16] = (_Float16)zn;
            }
        }
    }
}

extern "C" void kernel_launch(void* const* d_in, const int* in_sizes, int n_in,
                              void* d_out, int out_size, void* d_ws, size_t ws_size,
                              hipStream_t stream) {
    const float* x     = (const float*)d_in[0];
    const float* W_in  = (const float*)d_in[1];
    const float* W_rec = (const float*)d_in[2];
    float* out         = (float*)d_out;
    _Float16* Wp       = (_Float16*)d_ws;      // 640 KB packed weights

    int pack_elems = KTILES * NTILES * 512;    // 327,680
    pack_w_kernel<<<(pack_elems + 255) / 256, 256, 0, stream>>>(W_in, W_rec, Wp);
    alif_scan_kernel<<<256 / 16, 256, 0, stream>>>(x, Wp, out);
}
